// MambaClassifier_8641474200047
// MI455X (gfx1250) — matrix operand order repaired and checked by reading, not compiled, not run on hardware
//
#include <hip/hip_runtime.h>

// Mamba classifier: B=16, T=512, N=64, D=64, S=16, C=7
#define Tn 512
#define Nn 64
#define Dn 64
#define Sn 16
#define Bz 16
#define Cn 7
#define Rn (Bz * Nn * Tn) // 524288 rows of (BN*T, D)

typedef __bf16 bf16_t;
typedef __bf16 v16bf __attribute__((ext_vector_type(16)));
typedef float v8f __attribute__((ext_vector_type(8)));

union FragU {
  v16bf v;
  uint4 q[2];
};

// A-matrix fragment (16x32 bf16, row-major source). lane<16: K chunks [0..7],[16..23];
// lane>=16: [8..15],[24..31]. kBase in {0,32} selects which 32-wide K slice.
__device__ __forceinline__ v16bf frag_a(const bf16_t* row, int kBase, int lane) {
  int hi = (lane >> 4) << 3; // 0 or 8
  FragU f;
  f.q[0] = *(const uint4*)(row + kBase + hi);
  f.q[1] = *(const uint4*)(row + kBase + 16 + hi);
  return f.v;
}

// B-matrix fragment (32x16 bf16). Lane holds column n=lane&15; lanes<16 hold K chunks [0..7],[16..23];
// lanes>=16 hold [8..15],[24..31] (same as A). Source is row n of B^T (row-major).
__device__ __forceinline__ v16bf frag_b(const bf16_t* row, int kBase, int lane) {
  int hi = (lane >> 4) << 3; // 0 or 8
  FragU f;
  f.q[0] = *(const uint4*)(row + kBase + hi);
  f.q[1] = *(const uint4*)(row + kBase + 16 + hi);
  return f.v;
}

__device__ __forceinline__ v16bf frag_zero() {
  FragU f;
  f.q[0] = uint4{0u, 0u, 0u, 0u};
  f.q[1] = uint4{0u, 0u, 0u, 0u};
  return f.v;
}

__device__ __forceinline__ v8f wmma_bf16(v16bf a, v16bf b, v8f c) {
  return __builtin_amdgcn_wmma_f32_16x16x32_bf16(false, a, false, b, (short)0, c,
                                                 false, false);
}

// ---------------- zero init for pooled accumulator ----------------
__global__ void k_zero(float* __restrict__ p, int n) {
  int i = blockIdx.x * 256 + threadIdx.x;
  if (i < n) p[i] = 0.f;
}

// ---------------- LayerNorm: one wave per row, write bf16 ----------------
__global__ __launch_bounds__(256) void k_ln(const float* __restrict__ x,
                                            const float* __restrict__ w,
                                            const float* __restrict__ bvec,
                                            bf16_t* __restrict__ xn) {
  int wave = threadIdx.x >> 5, lane = threadIdx.x & 31;
  long r = (long)blockIdx.x * 8 + wave; // row in (BN*T)
  int bn = (int)(r >> 9);               // /T
  int t = (int)(r & (Tn - 1));
  int b = bn >> 6, n = bn & 63;
  const float* src = x + (((long)b * Tn + t) * Nn + n) * Dn;
  float v0 = src[lane], v1 = src[lane + 32];
  float s = v0 + v1;
#pragma unroll
  for (int m = 16; m >= 1; m >>= 1) s += __shfl_xor(s, m, 32);
  float mu = s * (1.0f / 64.0f);
  float d0 = v0 - mu, d1 = v1 - mu;
  float q = d0 * d0 + d1 * d1;
#pragma unroll
  for (int m = 16; m >= 1; m >>= 1) q += __shfl_xor(q, m, 32);
  float rs = rsqrtf(q * (1.0f / 64.0f) + 1e-5f);
  bf16_t* dst = xn + r * Dn;
  dst[lane] = (bf16_t)(d0 * rs * w[lane] + bvec[lane]);
  dst[lane + 32] = (bf16_t)(d1 * rs * w[lane + 32] + bvec[lane + 32]);
}

// ---------------- in_proj GEMM: (R,64)@(64,128) + bias, split x1(bf16)/x2(f32) --------
__global__ __launch_bounds__(256) void k_inproj(const bf16_t* __restrict__ xn,
                                                const float* __restrict__ W,
                                                const float* __restrict__ bias,
                                                bf16_t* __restrict__ x1,
                                                float* __restrict__ x2) {
  __shared__ bf16_t Wl[128][64]; // Wl[c][d] = W[c,d] (already B^T layout)
  for (int i = threadIdx.x; i < 128 * 64; i += 256) Wl[i >> 6][i & 63] = (bf16_t)W[i];
  __syncthreads();
  int wave = threadIdx.x >> 5, lane = threadIdx.x & 31;
  long r0 = (long)blockIdx.x * 16;
  int tileC = wave * 16;
  int mrow = lane & 15;
  const bf16_t* aRow = xn + (r0 + mrow) * Dn;
  v16bf a0 = frag_a(aRow, 0, lane);
  v16bf a1 = frag_a(aRow, 32, lane);
  const bf16_t* bRow = &Wl[tileC + mrow][0];
  v16bf b0 = frag_b(bRow, 0, lane);
  v16bf b1 = frag_b(bRow, 32, lane);
  v8f c = {0.f, 0.f, 0.f, 0.f, 0.f, 0.f, 0.f, 0.f};
  c = wmma_bf16(a0, b0, c);
  c = wmma_bf16(a1, b1, c);
  int nn = lane & 15, mB = (lane >> 4) << 3;
  int col = tileC + nn;
  float bv = bias[col];
  if (col < 64) {
#pragma unroll
    for (int j = 0; j < 8; ++j) x1[(r0 + mB + j) * Dn + col] = (bf16_t)(c[j] + bv);
  } else {
#pragma unroll
    for (int j = 0; j < 8; ++j) x2[(r0 + mB + j) * Dn + (col - 64)] = c[j] + bv;
  }
}

// ---------------- Conv1d(k=3,pad=1) as 3 accumulating GEMMs over shifted rows --------
__global__ __launch_bounds__(256) void k_conv(const bf16_t* __restrict__ x1,
                                              const float* __restrict__ cw,
                                              const float* __restrict__ cb,
                                              float* __restrict__ xc,
                                              bf16_t* __restrict__ xcb) {
  __shared__ bf16_t Wl[3][64][64]; // Wl[k][o][i] = cw[o,i,k]
  for (int i = threadIdx.x; i < 3 * 64 * 64; i += 256) {
    int k = i >> 12, rem = i & 4095, o = rem >> 6, ii = rem & 63;
    Wl[k][o][ii] = (bf16_t)cw[(o * 64 + ii) * 3 + k];
  }
  __syncthreads();
  int wave = threadIdx.x >> 5, lane = threadIdx.x & 31;
  int rowTile = blockIdx.x * 2 + (wave >> 2);
  int tileC = (wave & 3) * 16;
  int bn = rowTile >> 5; // 32 row-tiles per sequence
  int t0 = (rowTile & 31) << 4;
  long seqBase = (long)bn * Tn * Dn;
  int mrow = lane & 15;
  v8f c = {0.f, 0.f, 0.f, 0.f, 0.f, 0.f, 0.f, 0.f};
#pragma unroll
  for (int k = 0; k < 3; ++k) {
    int t = t0 + mrow + k - 1;
    v16bf a0, a1;
    if (t >= 0 && t < Tn) {
      const bf16_t* aRow = x1 + seqBase + (long)t * Dn;
      a0 = frag_a(aRow, 0, lane);
      a1 = frag_a(aRow, 32, lane);
    } else {
      a0 = frag_zero();
      a1 = frag_zero();
    }
    const bf16_t* bRow = &Wl[k][tileC + mrow][0];
    c = wmma_bf16(a0, frag_b(bRow, 0, lane), c);
    c = wmma_bf16(a1, frag_b(bRow, 32, lane), c);
  }
  int nn = lane & 15, mB = (lane >> 4) << 3;
  int col = tileC + nn;
  float bv = cb[col];
  long outBase = seqBase + (long)t0 * Dn;
#pragma unroll
  for (int j = 0; j < 8; ++j) {
    float v = c[j] + bv;
    xc[outBase + (mB + j) * Dn + col] = v;
    xcb[outBase + (mB + j) * Dn + col] = (bf16_t)v;
  }
}

// ---------------- dt/B/C projections: (R,64)@(64,96), softplus on dt cols ----------
__global__ __launch_bounds__(192) void k_proj(const bf16_t* __restrict__ xcb,
                                              const float* __restrict__ Wdt,
                                              const float* __restrict__ bdt,
                                              const float* __restrict__ WB,
                                              const float* __restrict__ WC,
                                              float* __restrict__ dtbc) {
  __shared__ bf16_t Wl[96][64]; // Wl[c][d]: c<64 -> Wdt[d,c]; 64..79 -> WB; 80..95 -> WC
  for (int i = threadIdx.x; i < 96 * 64; i += 192) {
    int cc = i >> 6, d = i & 63;
    float v = (cc < 64) ? Wdt[d * 64 + cc]
                        : ((cc < 80) ? WB[d * 16 + (cc - 64)] : WC[d * 16 + (cc - 80)]);
    Wl[cc][d] = (bf16_t)v;
  }
  __syncthreads();
  int wave = threadIdx.x >> 5, lane = threadIdx.x & 31;
  long r0 = (long)blockIdx.x * 16;
  int tileC = wave * 16;
  int mrow = lane & 15;
  const bf16_t* aRow = xcb + (r0 + mrow) * Dn;
  v16bf a0 = frag_a(aRow, 0, lane);
  v16bf a1 = frag_a(aRow, 32, lane);
  const bf16_t* bRow = &Wl[tileC + mrow][0];
  v8f c = {0.f, 0.f, 0.f, 0.f, 0.f, 0.f, 0.f, 0.f};
  c = wmma_bf16(a0, frag_b(bRow, 0, lane), c);
  c = wmma_bf16(a1, frag_b(bRow, 32, lane), c);
  int nn = lane & 15, mB = (lane >> 4) << 3;
  int col = tileC + nn;
#pragma unroll
  for (int j = 0; j < 8; ++j) {
    float v = c[j];
    if (col < 64) {
      v += bdt[col];
      v = (v > 20.f) ? v : __logf(1.f + __expf(v)); // softplus
    }
    dtbc[(r0 + mB + j) * 96 + col] = v;
  }
}

// ---------------- selective scan: one block per sequence, thread d holds h[d,0..15] ---
__global__ __launch_bounds__(64) void k_scan(const float* __restrict__ dtbc,
                                             const float* __restrict__ xc,
                                             const float* __restrict__ x2,
                                             const float* __restrict__ A_log,
                                             const float* __restrict__ Dskip,
                                             bf16_t* __restrict__ gated) {
  int bn = blockIdx.x;
  int d = threadIdx.x; // channel
  __shared__ float sDt[16][96]; // per-t: [0..63]=dt, [64..79]=B, [80..95]=C
  __shared__ float sX[16][64];
  __shared__ float sX2[16][64];
  float a[16], h[16];
#pragma unroll
  for (int s = 0; s < 16; ++s) {
    a[s] = -__expf(A_log[d * 16 + s]);
    h[s] = 0.f;
  }
  float dsk = Dskip[d];
  long rowBase = (long)bn * Tn;
  for (int c0 = 0; c0 < Tn; c0 += 16) {
    long dBase = (rowBase + c0) * 96;
    long xBase = (rowBase + c0) * 64;
    for (int i = d; i < 16 * 96; i += 64) ((float*)sDt)[i] = dtbc[dBase + i];
    for (int i = d; i < 16 * 64; i += 64) {
      ((float*)sX)[i] = xc[xBase + i];
      ((float*)sX2)[i] = x2[xBase + i];
    }
    __syncthreads();
    for (int tt = 0; tt < 16; ++tt) {
      float dt = sDt[tt][d];
      float xv = sX[tt][d];
      float g2 = sX2[tt][d];
      float dtx = dt * xv;
      float y = 0.f;
#pragma unroll
      for (int s = 0; s < 16; ++s) {
        h[s] = h[s] * __expf(dt * a[s]) + dtx * sDt[tt][64 + s];
        y += h[s] * sDt[tt][80 + s];
      }
      y += xv * dsk;                         // + x * D_skip
      float si = g2 / (1.f + __expf(-g2));   // silu(x2)
      float gate = 1.f / (1.f + __expf(-si)); // sigmoid(silu(x2))
      gated[(rowBase + c0 + tt) * 64 + d] = (bf16_t)(y * gate);
    }
    __syncthreads();
  }
}

// ---------------- out_proj GEMM + reduction over T, atomic into pooled ----------------
__global__ __launch_bounds__(256) void k_outproj(const bf16_t* __restrict__ gated,
                                                 const float* __restrict__ Wout,
                                                 float* __restrict__ pooled) {
  __shared__ bf16_t Wl[64][64]; // Wl[c][d] = Wout[c,d]
  __shared__ float colsum[64];
  for (int i = threadIdx.x; i < 64 * 64; i += 256) Wl[i >> 6][i & 63] = (bf16_t)Wout[i];
  if (threadIdx.x < 64) colsum[threadIdx.x] = 0.f;
  __syncthreads();
  int wave = threadIdx.x >> 5, lane = threadIdx.x & 31;
  int bn = blockIdx.x;
  int tileC = (wave & 3) * 16;
  int mrow = lane & 15;
  const bf16_t* bRow = &Wl[tileC + mrow][0];
  v16bf b0 = frag_b(bRow, 0, lane);
  v16bf b1 = frag_b(bRow, 32, lane);
  float acc = 0.f;
  for (int rt = (wave >> 2); rt < 32; rt += 2) {
    const bf16_t* aRow = gated + ((long)bn * Tn + rt * 16 + mrow) * 64;
    v16bf a0 = frag_a(aRow, 0, lane);
    v16bf a1 = frag_a(aRow, 32, lane);
    v8f c = {0.f, 0.f, 0.f, 0.f, 0.f, 0.f, 0.f, 0.f};
    c = wmma_bf16(a0, b0, c);
    c = wmma_bf16(a1, b1, c);
#pragma unroll
    for (int j = 0; j < 8; ++j) acc += c[j]; // sum over the 8 rows this lane owns
  }
  atomicAdd(&colsum[tileC + (lane & 15)], acc); // ds_add_f32
  __syncthreads();
  if (threadIdx.x < 64) {
    int b = bn >> 6;
    atomicAdd(&pooled[b * 64 + threadIdx.x],
              colsum[threadIdx.x] * (1.0f / ((float)Tn * (float)Nn)));
  }
}

// ---------------- residual mean over (t, n) ----------------
__global__ __launch_bounds__(256) void k_resid(const float* __restrict__ x,
                                               float* __restrict__ pooled) {
  __shared__ float sAcc[256];
  int b = blockIdx.x >> 4;
  int chunk = blockIdx.x & 15; // 32 t-steps per chunk
  long base = ((long)b * Tn + (long)chunk * 32) * (Nn * Dn);
  float acc = 0.f;
  for (int i = threadIdx.x; i < 32 * Nn * Dn; i += 256) acc += x[base + i];
  sAcc[threadIdx.x] = acc; // d = threadIdx.x & 63 is invariant across the loop
  __syncthreads();
  if (threadIdx.x < 64) {
    float s = sAcc[threadIdx.x] + sAcc[threadIdx.x + 64] + sAcc[threadIdx.x + 128] +
              sAcc[threadIdx.x + 192];
    atomicAdd(&pooled[b * 64 + threadIdx.x], s * (1.0f / ((float)Tn * (float)Nn)));
  }
}

// ---------------- classifier: (B,64)@(64,7)^T ----------------
__global__ __launch_bounds__(256) void k_final(const float* __restrict__ pooled,
                                               const float* __restrict__ outb,
                                               const float* __restrict__ clsw,
                                               const float* __restrict__ clsb,
                                               float* __restrict__ out) {
  __shared__ float P[Bz * 64];
  for (int i = threadIdx.x; i < Bz * 64; i += 256) P[i] = pooled[i] + outb[i & 63];
  __syncthreads();
  if (threadIdx.x < Bz * Cn) {
    int b = threadIdx.x / Cn, cc = threadIdx.x % Cn;
    float s = clsb[cc];
#pragma unroll 8
    for (int d = 0; d < 64; ++d) s += P[b * 64 + d] * clsw[cc * 64 + d];
    out[threadIdx.x] = s;
  }
}

extern "C" void kernel_launch(void* const* d_in, const int* in_sizes, int n_in,
                              void* d_out, int out_size, void* d_ws, size_t ws_size,
                              hipStream_t stream) {
  (void)in_sizes; (void)n_in; (void)out_size; (void)ws_size;
  const float* x = (const float*)d_in[0];
  const float* norm_w = (const float*)d_in[1];
  const float* norm_b = (const float*)d_in[2];
  const float* inW = (const float*)d_in[3];
  const float* inB = (const float*)d_in[4];
  const float* convW = (const float*)d_in[5];
  const float* convB = (const float*)d_in[6];
  const float* A_log = (const float*)d_in[7];
  const float* W_dt = (const float*)d_in[8];
  const float* b_dt = (const float*)d_in[9];
  const float* W_B = (const float*)d_in[10];
  const float* W_C = (const float*)d_in[11];
  const float* Dskip = (const float*)d_in[12];
  const float* outW = (const float*)d_in[13];
  const float* outB = (const float*)d_in[14];
  const float* clsW = (const float*)d_in[15];
  const float* clsB = (const float*)d_in[16];

  char* ws = (char*)d_ws;
  bf16_t* xn = (bf16_t*)(ws + (size_t)0);           // 64 MB  normalized (bf16)
  bf16_t* x1 = (bf16_t*)(ws + (size_t)67108864);    // 64 MB  in_proj half 1 (bf16)
  float* x2 = (float*)(ws + (size_t)134217728);     // 128 MB in_proj half 2 (f32)
  float* xc = (float*)(ws + (size_t)268435456);     // 128 MB conv out (f32)
  bf16_t* xcb = (bf16_t*)(ws + (size_t)402653184);  // 64 MB  conv out (bf16)
  float* dtbc = (float*)(ws + (size_t)469762048);   // 192 MB dt|B|C (f32, 96/row)
  bf16_t* gated = (bf16_t*)(ws + (size_t)671088640); // 64 MB gated (bf16)
  float* pooled = (float*)(ws + (size_t)738197504); // 4 KB pooled accum

  k_zero<<<4, 256, 0, stream>>>(pooled, Bz * 64);
  k_ln<<<Rn / 8, 256, 0, stream>>>(x, norm_w, norm_b, xn);
  k_inproj<<<Rn / 16, 256, 0, stream>>>(xn, inW, inB, x1, x2);
  k_conv<<<Rn / 32, 256, 0, stream>>>(x1, convW, convB, xc, xcb);
  k_proj<<<Rn / 16, 192, 0, stream>>>(xcb, W_dt, b_dt, W_B, W_C, dtbc);
  k_scan<<<Bz * Nn, 64, 0, stream>>>(dtbc, xc, x2, A_log, Dskip, gated);
  k_outproj<<<Bz * Nn, 256, 0, stream>>>(gated, outW, pooled);
  k_resid<<<Bz * 16, 256, 0, stream>>>(x, pooled);
  k_final<<<1, 256, 0, stream>>>(pooled, outB, clsW, clsB, (float*)d_out);
}
